// Upscale_82042465289119
// MI455X (gfx1250) — matrix operand order repaired and checked by reading, not compiled, not run on hardware
//
#include <hip/hip_runtime.h>

// Problem constants (from reference)
#define M_PTS   35637
#define K_NB    20
#define FEAT    128
#define P_TOTAL (M_PTS * K_NB)                      // 712740 points
#define TILE_P  64
#define NBLK    ((P_TOTAL + TILE_P - 1) / TILE_P)   // 11137 workgroups per GEMM layer
#define EPS_BN  1e-5f
#define W_ELEMS (7 * FEAT * FEAT)                   // all conv weights (layers 1..7)

typedef __attribute__((ext_vector_type(16))) __bf16 v16bf;
typedef __attribute__((ext_vector_type(8)))  float  v8f;

union FragBF {
  v16bf v;
  struct { uint4 lo; uint4 hi; } q;
};
union Pack8 { uint4 q; __bf16 e[8]; };
union Pack4 { uint2 q; __bf16 e[4]; };

// ---------------------------------------------------------------------------
// Zero the per-layer (sum, sumsq) stats accumulators (8 layers x 256 floats).
// ---------------------------------------------------------------------------
__global__ void zero_stats_kernel(float* __restrict__ sums) {
  int i = blockIdx.x * blockDim.x + threadIdx.x;
  if (i < 8 * 256) sums[i] = 0.0f;
}

// ---------------------------------------------------------------------------
// One-shot: convert all conv weights (7 x 128 x 128 f32) to bf16 in workspace.
// ---------------------------------------------------------------------------
__global__ void convert_w_kernel(const float* __restrict__ Wf,
                                 __bf16* __restrict__ Wb) {
  const int i = blockIdx.x * blockDim.x + threadIdx.x;   // float4 chunk index
  if (i < W_ELEMS / 4) {
    const float4 w4 = ((const float4*)Wf)[i];
    Pack4 p4;
    p4.e[0] = (__bf16)w4.x; p4.e[1] = (__bf16)w4.y;
    p4.e[2] = (__bf16)w4.z; p4.e[3] = (__bf16)w4.w;
    *(uint2*)(Wb + i * 4) = p4.q;
  }
}

// ---------------------------------------------------------------------------
// Layer 0: z0[p,o] = sum_{c<7} w_up[c,p] * W0[o,c]   (7-wide, VALU is fine)
// Also accumulates per-channel sum / sumsq for the BN stats of layer 0.
// blockDim = 128 (one thread per output channel), each block does 64 points.
// ---------------------------------------------------------------------------
__global__ void layer0_kernel(const float* __restrict__ in,   // (7, P)
                              const float* __restrict__ W0,   // (128, 7)
                              __bf16* __restrict__ zb,        // (P_pad, 128) bf16
                              float* __restrict__ sums) {     // [sum(128), sumsq(128)]
  const int o = threadIdx.x;
  const long p0 = (long)blockIdx.x * 64;
  float w[7];
#pragma unroll
  for (int c = 0; c < 7; ++c) w[c] = W0[o * 7 + c];
  float s = 0.0f, q = 0.0f;
  for (int j = 0; j < 64; ++j) {
    const long p = p0 + j;
    if (p >= P_TOTAL) break;
    float z = 0.0f;
#pragma unroll
    for (int c = 0; c < 7; ++c) z = fmaf(in[(long)c * P_TOTAL + p], w[c], z);
    zb[p * FEAT + o] = (__bf16)z;
    s += z;
    q += z * z;
  }
  atomicAdd(&sums[o], s);
  atomicAdd(&sums[FEAT + o], q);
}

// ---------------------------------------------------------------------------
// Turn (sum, sumsq, gamma, beta) into the folded affine (a, c):
//   h = sin(a * z + c),  a = g * rsqrt(var+eps),  c = b - mu * a
// ---------------------------------------------------------------------------
__global__ void finalize_kernel(const float* __restrict__ sums,
                                const float* __restrict__ g,
                                const float* __restrict__ b,
                                float* __restrict__ ac) {
  const int o = threadIdx.x;  // 128 threads
  const float invP = 1.0f / (float)P_TOTAL;
  const float mu  = sums[o] * invP;
  const float var = sums[FEAT + o] * invP - mu * mu;
  const float a   = g[o] * rsqrtf(var + EPS_BN);
  ac[o] = a;
  ac[FEAT + o] = b[o] - mu * a;
}

// ---------------------------------------------------------------------------
// Fused layer L (L = 1..7):
//   h = sin(aPrev * zPrev + cPrev)            (applied while staging to LDS)
//   zNew[p,o] = sum_c h[p,c] * W[o,c]         (WMMA bf16, f32 accumulate)
//   in-place overwrite of the z buffer tile; per-channel stats accumulated.
// 256 threads = 8 wave32; each wave owns a 16(points) x 64(channels) tile.
// ---------------------------------------------------------------------------
__global__ __launch_bounds__(256) void gemm_bn_sine_kernel(
    __bf16* __restrict__ zb,             // (P_pad, 128) bf16, read+write in place
    const __bf16* __restrict__ Wb,       // (128, 128) bf16, W[o][c] (pre-converted)
    const float* __restrict__ acPrev,    // [a(128), c(128)] of producing layer
    float* __restrict__ sumsOut) {       // [sum(128), sumsq(128)] of this layer
  __shared__ __bf16 hTile[TILE_P * FEAT];  // 16 KB
  __shared__ __bf16 wTile[FEAT * FEAT];    // 32 KB
  __shared__ float  sA[FEAT], sC[FEAT], redS[FEAT], redQ[FEAT];

  const int  tid   = threadIdx.x;
  const long pBase = (long)blockIdx.x * TILE_P;

  // Kick the z tile toward the caches while W is being staged.
  __builtin_prefetch((const char*)(zb + pBase * FEAT) + tid * 64, 0, 0);

  if (tid < FEAT) {
    sA[tid]   = acPrev[tid];
    sC[tid]   = acPrev[FEAT + tid];
    redS[tid] = 0.0f;
    redQ[tid] = 0.0f;
  }
  // Stage W (bf16) into LDS with plain 16B copies.
  for (int i = tid; i < FEAT * FEAT / 8; i += 256) {
    *(uint4*)&wTile[i * 8] = ((const uint4*)Wb)[i];
  }
  __syncthreads();

  // Stage h = sin(a*z + c) tile into LDS (bf16), 16B vectors.
  for (int i = tid; i < TILE_P * FEAT / 8; i += 256) {
    const int base = i << 3;
    const int pl   = base >> 7;
    const int ch   = base & (FEAT - 1);
    Pack8 in8, out8;
    in8.q = *(const uint4*)(zb + (pBase + pl) * FEAT + ch);
#pragma unroll
    for (int j = 0; j < 8; ++j) {
      const float z = (float)in8.e[j];
      out8.e[j] = (__bf16)__sinf(fmaf(sA[ch + j], z, sC[ch + j]));
    }
    *(uint4*)&hTile[base] = out8.q;
  }
  __syncthreads();

  const int wv     = tid >> 5;
  const int ln     = tid & 31;
  const int laneLo = ln & 15;
  const int hiSel  = ln >> 4;
  const int mBase  = (wv >> 1) * 16;   // 4 point-groups of 16 rows
  const int nBase  = (wv & 1) * 4;     // 2 halves of the 8 N-tiles

  v8f acc[4];
#pragma unroll
  for (int nt = 0; nt < 4; ++nt) { v8f z = {}; acc[nt] = z; }

#pragma unroll
  for (int kk = 0; kk < 4; ++kk) {
    // A fragment (16x32 bf16): lane holds K = {kb..kb+15}
    const int kbA = kk * 32 + hiSel * 16;
    FragBF a;
    a.q.lo = *(const uint4*)&hTile[(mBase + laneLo) * FEAT + kbA];
    a.q.hi = *(const uint4*)&hTile[(mBase + laneLo) * FEAT + kbA + 8];
#pragma unroll
    for (int nt = 0; nt < 4; ++nt) {
      const int o = (nBase + nt) * 16 + laneLo;
      // B fragment (32x16 bf16): lane-half selects contiguous 16 K-values
      const __bf16* bp = &wTile[o * FEAT + kk * 32 + hiSel * 16];
      FragBF bf;
      bf.q.lo = *(const uint4*)bp;
      bf.q.hi = *(const uint4*)(bp + 8);
      acc[nt] = __builtin_amdgcn_wmma_f32_16x16x32_bf16(
          false, a.v, false, bf.v, (short)0, acc[nt], false, false);
    }
  }

  // Write z back (bf16, in place) + per-channel stats (masked for tail rows).
#pragma unroll
  for (int nt = 0; nt < 4; ++nt) {
    const int o = (nBase + nt) * 16 + laneLo;
    float s = 0.0f, q = 0.0f;
#pragma unroll
    for (int r = 0; r < 8; ++r) {
      const int  m = hiSel ? (8 + r) : r;  // C/D layout: VGPR r -> row r or 8+r
      const long p = pBase + mBase + m;
      const float zv = acc[nt][r];
      if (p < P_TOTAL) {
        zb[p * FEAT + o] = (__bf16)zv;
        s += zv;
        q += zv * zv;
      }
    }
    atomicAdd(&redS[o], s);
    atomicAdd(&redQ[o], q);
  }
  __syncthreads();
  if (tid < FEAT) {
    atomicAdd(&sumsOut[tid], redS[tid]);
    atomicAdd(&sumsOut[FEAT + tid], redQ[tid]);
  }
}

// ---------------------------------------------------------------------------
// Head: logits[k] = dot(sin(a7*z7 + c7), final_w) + b; softmax over K=20;
// out[m,:] = sum_k softmax_k * x_feat[idx[m,k], :].
// One wave32 per point m; each lane owns 4 of the 128 feature channels.
// ---------------------------------------------------------------------------
__global__ __launch_bounds__(256) void final_kernel(
    const __bf16* __restrict__ zb, const float* __restrict__ ac7,
    const float* __restrict__ fw, const float* __restrict__ fb,
    const int* __restrict__ idx, const float* __restrict__ xfeat,
    float* __restrict__ out) {
  const int tid = threadIdx.x;
  const int wv  = tid >> 5;
  const int ln  = tid & 31;
  const long m  = (long)blockIdx.x * 8 + wv;
  if (m >= M_PTS) return;
  const int ch0 = ln * 4;

  float a4[4], c4[4], w4[4];
#pragma unroll
  for (int j = 0; j < 4; ++j) {
    a4[j] = ac7[ch0 + j];
    c4[j] = ac7[FEAT + ch0 + j];
    w4[j] = fw[ch0 + j];
  }

  float logit[K_NB];
#pragma unroll
  for (int k = 0; k < K_NB; ++k) {
    const long p = m * K_NB + k;
    float part = 0.0f;
#pragma unroll
    for (int j = 0; j < 4; ++j) {
      const float z = (float)zb[p * FEAT + ch0 + j];
      part += __sinf(fmaf(a4[j], z, c4[j])) * w4[j];
    }
#pragma unroll
    for (int off = 16; off > 0; off >>= 1)
      part += __shfl_xor(part, off, 32);
    logit[k] = part + fb[0];
  }

  float mx = logit[0];
#pragma unroll
  for (int k = 1; k < K_NB; ++k) mx = fmaxf(mx, logit[k]);
  float e[K_NB], s = 0.0f;
#pragma unroll
  for (int k = 0; k < K_NB; ++k) { e[k] = __expf(logit[k] - mx); s += e[k]; }
  const float inv = 1.0f / s;

  float acc0 = 0.f, acc1 = 0.f, acc2 = 0.f, acc3 = 0.f;
#pragma unroll
  for (int k = 0; k < K_NB; ++k) {
    const long g  = idx[m * K_NB + k];
    const float wk = e[k] * inv;
    const float4 xv = *(const float4*)(xfeat + g * FEAT + ch0);  // hits L2 (8 MB table)
    acc0 = fmaf(wk, xv.x, acc0);
    acc1 = fmaf(wk, xv.y, acc1);
    acc2 = fmaf(wk, xv.z, acc2);
    acc3 = fmaf(wk, xv.w, acc3);
  }
  *(float4*)(out + m * FEAT + ch0) = make_float4(acc0, acc1, acc2, acc3);
}

// ---------------------------------------------------------------------------
// Workspace layout:
//   [sums: 8*256 f32][ac: 8*256 f32][Wb: 7*128*128 bf16][z buffer: P_pad*128 bf16]
// Required ws_size ~ 175 MB.
// ---------------------------------------------------------------------------
extern "C" void kernel_launch(void* const* d_in, const int* in_sizes, int n_in,
                              void* d_out, int out_size, void* d_ws, size_t ws_size,
                              hipStream_t stream) {
  (void)in_sizes; (void)n_in; (void)out_size; (void)ws_size;
  const float* x_feat  = (const float*)d_in[0];
  const float* w_up    = (const float*)d_in[1];
  const int*   idx     = (const int*)d_in[2];
  const float* conv0_w = (const float*)d_in[3];
  const float* convs_w = (const float*)d_in[4];
  const float* gammas  = (const float*)d_in[5];
  const float* betas   = (const float*)d_in[6];
  const float* final_w = (const float*)d_in[7];
  const float* final_b = (const float*)d_in[8];
  float* out = (float*)d_out;

  char*   ws   = (char*)d_ws;
  float*  sums = (float*)ws;                         // 8 x (sum[128], sumsq[128])
  float*  ac   = sums + 8 * 256;                     // 8 x (a[128], c[128])
  __bf16* Wb   = (__bf16*)(ws + 16384);              // 7 x 128 x 128 bf16 weights
  __bf16* zb   = (__bf16*)(ws + 16384 + W_ELEMS * 2);// padded z buffer

  zero_stats_kernel<<<8, 256, 0, stream>>>(sums);
  convert_w_kernel<<<(W_ELEMS / 4 + 255) / 256, 256, 0, stream>>>(convs_w, Wb);
  layer0_kernel<<<(P_TOTAL + 63) / 64, 128, 0, stream>>>(w_up, conv0_w, zb, sums);
  for (int L = 1; L <= 7; ++L) {
    finalize_kernel<<<1, 128, 0, stream>>>(sums + (L - 1) * 256,
                                           gammas + (L - 1) * 128,
                                           betas + (L - 1) * 128,
                                           ac + (L - 1) * 256);
    gemm_bn_sine_kernel<<<NBLK, 256, 0, stream>>>(zb,
                                                  Wb + (size_t)(L - 1) * FEAT * FEAT,
                                                  ac + (L - 1) * 256,
                                                  sums + L * 256);
  }
  finalize_kernel<<<1, 128, 0, stream>>>(sums + 7 * 256, gammas + 7 * 128,
                                         betas + 7 * 128, ac + 7 * 256);
  final_kernel<<<(M_PTS + 7) / 8, 256, 0, stream>>>(zb, ac + 7 * 256, final_w,
                                                    final_b, idx, x_feat, out);
}
